// MultiheadAttention_68942815035775
// MI455X (gfx1250) — matrix operand order repaired and checked by reading, not compiled, not run on hardware
//
#include <hip/hip_runtime.h>

// ---------------------------------------------------------------------------
// Multihead attention forward, MI455X (gfx1250), bf16 WMMA + fp32 accumulate.
// T=2048, B=2, E=1024, H=16, D=64.  N = T*B = 4096 rows.
// Pipeline:
//   1) pack: fp32 -> bf16 for activations + 4 weight matrices
//   2) gemm_qkv (x3): X @ W.T + b, Q scaled by D^-0.5; Q,K -> [B,H,T,D],
//      V -> transposed [B,H,D,T] so P*V WMMA B-frags are contiguous
//   3) flash_attn: per (head, 32-query tile) wave; K/V tiles double-buffered
//      into LDS with global_load_async_to_lds_b128 (ASYNCcnt), online softmax
//      over 32-key chunks, S=Q*K^T and O+=P*V on v_wmma_f32_16x16x32_bf16
//   4) gemm_out: ctx @ Wo.T + b -> fp32 output
// ---------------------------------------------------------------------------

#define T_SEQ  2048
#define BATCH  2
#define EMB    1024
#define HEADS  16
#define DHEAD  64
#define NROWS  (T_SEQ * BATCH)      // 4096

typedef __attribute__((ext_vector_type(16))) __bf16       v16bf;
typedef __attribute__((ext_vector_type(8)))  float        v8f;
typedef unsigned int u32;
typedef __attribute__((ext_vector_type(4)))  u32          u32x4;

union FragAB {            // one 16x32 (A) or 32x16 (B) bf16 fragment: 32B/lane
    v16bf v;
    u32x4 u[2];
};

__device__ __forceinline__ unsigned short f2bf(float f) {
    union { float f; u32 u; } c; c.f = f;
    u32 u = c.u;
    u += 0x7FFFu + ((u >> 16) & 1u);      // round-to-nearest-even
    return (unsigned short)(u >> 16);
}

// Async global -> LDS copy of 16 bytes (tracked by ASYNCcnt).
// lds_off = byte address within LDS (low 32 bits of the flat shared address).
__device__ __forceinline__ void async_copy16(unsigned lds_off,
                                             const void* gptr) {
    asm volatile("global_load_async_to_lds_b128 %0, %1, off"
                 :: "v"(lds_off), "v"(gptr)
                 : "memory");
}

__device__ __forceinline__ void wait_async0() {
    asm volatile("s_wait_asynccnt 0x0" ::: "memory");
}

// ---------------------------------------------------------------------------
// fp32 -> bf16 elementwise pack
// ---------------------------------------------------------------------------
__global__ void pack_bf16(const float* __restrict__ src,
                          unsigned short* __restrict__ dst, int n) {
    int i = blockIdx.x * blockDim.x + threadIdx.x;
    if (i < n) dst[i] = f2bf(src[i]);
}

// ---------------------------------------------------------------------------
// QKV projection: out = X @ W.T + bias   (one wave = 32x64 output macro-tile)
// mode 0: Q (scale by 0.125, store [B,H,T,D])
// mode 1: K (store [B,H,T,D])
// mode 2: V (store transposed [B,H,D,T])
// ---------------------------------------------------------------------------
__global__ __launch_bounds__(32)
void gemm_qkv(const unsigned short* __restrict__ X,
              const unsigned short* __restrict__ W,
              const float* __restrict__ bias,
              unsigned short* __restrict__ out, int mode) {
    const int lane = threadIdx.x & 31;
    const int lo = lane & 15, hi = lane >> 4;
    const int m0 = blockIdx.x * 32;
    const int n0 = blockIdx.y * 64;

    float bv[4];
#pragma unroll
    for (int nt = 0; nt < 4; ++nt) bv[nt] = bias[n0 + nt * 16 + lo];

    v8f acc[2][4];
#pragma unroll
    for (int mt = 0; mt < 2; ++mt)
#pragma unroll
        for (int nt = 0; nt < 4; ++nt) acc[mt][nt] = (v8f){};

    for (int kk = 0; kk < EMB; kk += 32) {
        FragAB a[2];
#pragma unroll
        for (int mt = 0; mt < 2; ++mt) {
            const unsigned short* p =
                X + (size_t)(m0 + mt * 16 + lo) * EMB + kk + hi * 8;
            a[mt].u[0] = *(const u32x4*)p;
            a[mt].u[1] = *(const u32x4*)(p + 16);
        }
        FragAB b[4];
#pragma unroll
        for (int nt = 0; nt < 4; ++nt) {
            const unsigned short* p =
                W + (size_t)(n0 + nt * 16 + lo) * EMB + kk + hi * 8;
            b[nt].u[0] = *(const u32x4*)p;
            b[nt].u[1] = *(const u32x4*)(p + 16);
        }
#pragma unroll
        for (int mt = 0; mt < 2; ++mt)
#pragma unroll
            for (int nt = 0; nt < 4; ++nt)
                acc[mt][nt] = __builtin_amdgcn_wmma_f32_16x16x32_bf16(
                    false, a[mt].v, false, b[nt].v, (short)0, acc[mt][nt],
                    false, false);
    }

    const float qscale = (mode == 0) ? 0.125f : 1.0f;   // DHEAD^-0.5
#pragma unroll
    for (int mt = 0; mt < 2; ++mt)
#pragma unroll
        for (int nt = 0; nt < 4; ++nt)
#pragma unroll
            for (int r = 0; r < 8; ++r) {
                int row = m0 + mt * 16 + hi * 8 + r;   // n = t*BATCH + b
                int e   = n0 + nt * 16 + lo;
                float v = (acc[mt][nt][r] + bv[nt]) * qscale;
                int t = row >> 1, bb = row & 1;        // BATCH == 2
                int h = e >> 6,  d  = e & 63;          // DHEAD == 64
                size_t dst;
                if (mode == 2)
                    dst = ((size_t)(bb * HEADS + h) * DHEAD + d) * T_SEQ + t;
                else
                    dst = ((size_t)(bb * HEADS + h) * T_SEQ + t) * DHEAD + d;
                out[dst] = f2bf(v);
            }
}

// ---------------------------------------------------------------------------
// Flash attention: one wave per (32-query tile, head). K/V tiles stream into
// LDS via double-buffered async copies; online softmax over 32-key chunks;
// P transposed C-layout -> A-layout via a 2 KB LDS tile.
// ---------------------------------------------------------------------------
__global__ __launch_bounds__(32)
void flash_attn(const unsigned short* __restrict__ Qp,
                const unsigned short* __restrict__ Kp,
                const unsigned short* __restrict__ Vt,
                unsigned short* __restrict__ ctx) {
    __shared__ unsigned short ldsP[32 * 32];        // P tile (2 KB)
    __shared__ unsigned short ldsK[2][32 * DHEAD];  // K dbl-buffer (2 x 4 KB)
    __shared__ unsigned short ldsV[2][DHEAD * 32];  // V dbl-buffer (2 x 4 KB)

    const int lane = threadIdx.x & 31;
    const int lo = lane & 15, hi = lane >> 4;
    const int q0 = blockIdx.x * 32;
    const int bh = blockIdx.y;                 // b*HEADS + h
    const int hh = bh & (HEADS - 1);
    const int bb = bh >> 4;

    const unsigned short* Qb = Qp + (size_t)bh * T_SEQ * DHEAD;
    const unsigned short* Kb = Kp + (size_t)bh * T_SEQ * DHEAD;
    const unsigned short* Vb = Vt + (size_t)bh * DHEAD * T_SEQ;

    const unsigned kofs[2] = { (unsigned)(uintptr_t)&ldsK[0][0],
                               (unsigned)(uintptr_t)&ldsK[1][0] };
    const unsigned vofs[2] = { (unsigned)(uintptr_t)&ldsV[0][0],
                               (unsigned)(uintptr_t)&ldsV[1][0] };

    // Stage one 32-key chunk: K tile is one contiguous 4 KB block
    // (rows of [T,D] storage), V tile is 64 rows x 64 B (stride T_SEQ).
    auto stage_kv = [&](int kv, int buf) {
        const char* gk = (const char*)(Kb + (size_t)kv * DHEAD);
#pragma unroll
        for (int i = 0; i < 8; ++i) {
            int seg = i * 32 + lane;                     // 256 x 16 B
            async_copy16(kofs[buf] + seg * 16, gk + seg * 16);
        }
        const char* gv = (const char*)(Vb + kv);
#pragma unroll
        for (int i = 0; i < 8; ++i) {
            int seg = i * 32 + lane;                     // d = seg>>2
            async_copy16(vofs[buf] + seg * 16,
                         gv + (size_t)(seg >> 2) * (T_SEQ * 2) +
                             (seg & 3) * 16);
        }
    };

    // Persistent Q fragments: 2 M-tiles x 2 K-steps over D=64
    FragAB qf[2][2];
#pragma unroll
    for (int mt = 0; mt < 2; ++mt)
#pragma unroll
        for (int kd = 0; kd < 2; ++kd) {
            const unsigned short* p =
                Qb + (size_t)(q0 + mt * 16 + lo) * DHEAD + kd * 32 + hi * 8;
            qf[mt][kd].u[0] = *(const u32x4*)p;
            qf[mt][kd].u[1] = *(const u32x4*)(p + 16);
        }

    v8f o[2][4];
    float m[2][8], l[2][8];
#pragma unroll
    for (int mt = 0; mt < 2; ++mt) {
#pragma unroll
        for (int dt = 0; dt < 4; ++dt) o[mt][dt] = (v8f){};
#pragma unroll
        for (int r = 0; r < 8; ++r) { m[mt][r] = -1e30f; l[mt][r] = 0.f; }
    }

    stage_kv(0, 0);          // prologue: chunk 0 into buffer 0
    wait_async0();

    for (int kv = 0; kv < T_SEQ; kv += 32) {
        const int buf = (kv >> 5) & 1;
        if (kv + 32 < T_SEQ) stage_kv(kv + 32, buf ^ 1);   // overlap next

        // ---- S = Q * K^T from LDS ---------------------------------------
        v8f s[2][2];
#pragma unroll
        for (int mt = 0; mt < 2; ++mt)
#pragma unroll
            for (int nt = 0; nt < 2; ++nt) s[mt][nt] = (v8f){};

        FragAB kf[2][2];
#pragma unroll
        for (int nt = 0; nt < 2; ++nt)
#pragma unroll
            for (int kd = 0; kd < 2; ++kd) {
                const unsigned short* p =
                    &ldsK[buf][(nt * 16 + lo) * DHEAD + kd * 32 + hi * 8];
                kf[nt][kd].u[0] = *(const u32x4*)p;
                kf[nt][kd].u[1] = *(const u32x4*)(p + 16);
            }
#pragma unroll
        for (int mt = 0; mt < 2; ++mt)
#pragma unroll
            for (int nt = 0; nt < 2; ++nt)
#pragma unroll
                for (int kd = 0; kd < 2; ++kd)
                    s[mt][nt] = __builtin_amdgcn_wmma_f32_16x16x32_bf16(
                        false, qf[mt][kd].v, false, kf[nt][kd].v, (short)0,
                        s[mt][nt], false, false);

        // ---- online softmax (row stats via 16-lane xor reductions) ------
#pragma unroll
        for (int mt = 0; mt < 2; ++mt)
#pragma unroll
            for (int r = 0; r < 8; ++r) {
                float vmax = fmaxf(s[mt][0][r], s[mt][1][r]);
#pragma unroll
                for (int xm = 1; xm < 16; xm <<= 1)
                    vmax = fmaxf(vmax, __shfl_xor(vmax, xm, 32));
                float mnew = fmaxf(m[mt][r], vmax);
                float sc = __expf(m[mt][r] - mnew);
                m[mt][r] = mnew;
                float p0 = __expf(s[mt][0][r] - mnew);
                float p1 = __expf(s[mt][1][r] - mnew);
                float ps = p0 + p1;
#pragma unroll
                for (int xm = 1; xm < 16; xm <<= 1)
                    ps += __shfl_xor(ps, xm, 32);
                l[mt][r] = l[mt][r] * sc + ps;
#pragma unroll
                for (int dt = 0; dt < 4; ++dt) o[mt][dt][r] *= sc;
                int row = mt * 16 + hi * 8 + r;          // C-layout row
                ldsP[row * 32 + lo]      = f2bf(p0);
                ldsP[row * 32 + 16 + lo] = f2bf(p1);
            }

        // ---- O += P * V (P from LDS in A-layout, V tile from LDS) --------
        FragAB pf[2];
#pragma unroll
        for (int mt = 0; mt < 2; ++mt) {
            const unsigned short* p = &ldsP[(mt * 16 + lo) * 32 + hi * 8];
            pf[mt].u[0] = *(const u32x4*)p;
            pf[mt].u[1] = *(const u32x4*)(p + 16);
        }
        FragAB vf[4];
#pragma unroll
        for (int dt = 0; dt < 4; ++dt) {
            const unsigned short* p =
                &ldsV[buf][(dt * 16 + lo) * 32 + hi * 8];
            vf[dt].u[0] = *(const u32x4*)p;
            vf[dt].u[1] = *(const u32x4*)(p + 16);
        }
#pragma unroll
        for (int mt = 0; mt < 2; ++mt)
#pragma unroll
            for (int dt = 0; dt < 4; ++dt)
                o[mt][dt] = __builtin_amdgcn_wmma_f32_16x16x32_bf16(
                    false, pf[mt].v, false, vf[dt].v, (short)0, o[mt][dt],
                    false, false);

        wait_async0();      // next chunk's async copies complete
    }

    // ---- normalize + scatter to ctx[T,B,E] in bf16 ----------------------
#pragma unroll
    for (int mt = 0; mt < 2; ++mt)
#pragma unroll
        for (int dt = 0; dt < 4; ++dt)
#pragma unroll
            for (int r = 0; r < 8; ++r) {
                float val = o[mt][dt][r] / l[mt][r];
                int t = q0 + mt * 16 + hi * 8 + r;
                int e = hh * DHEAD + dt * 16 + lo;
                ctx[(size_t)(t * BATCH + bb) * EMB + e] = f2bf(val);
            }
}

// ---------------------------------------------------------------------------
// Output projection: out = ctx @ Wo.T + bias   (fp32 result)
// ---------------------------------------------------------------------------
__global__ __launch_bounds__(32)
void gemm_out(const unsigned short* __restrict__ Xc,
              const unsigned short* __restrict__ W,
              const float* __restrict__ bias,
              float* __restrict__ out) {
    const int lane = threadIdx.x & 31;
    const int lo = lane & 15, hi = lane >> 4;
    const int m0 = blockIdx.x * 32;
    const int n0 = blockIdx.y * 64;

    float bv[4];
#pragma unroll
    for (int nt = 0; nt < 4; ++nt) bv[nt] = bias[n0 + nt * 16 + lo];

    v8f acc[2][4];
#pragma unroll
    for (int mt = 0; mt < 2; ++mt)
#pragma unroll
        for (int nt = 0; nt < 4; ++nt) acc[mt][nt] = (v8f){};

    for (int kk = 0; kk < EMB; kk += 32) {
        FragAB a[2];
#pragma unroll
        for (int mt = 0; mt < 2; ++mt) {
            const unsigned short* p =
                Xc + (size_t)(m0 + mt * 16 + lo) * EMB + kk + hi * 8;
            a[mt].u[0] = *(const u32x4*)p;
            a[mt].u[1] = *(const u32x4*)(p + 16);
        }
        FragAB b[4];
#pragma unroll
        for (int nt = 0; nt < 4; ++nt) {
            const unsigned short* p =
                W + (size_t)(n0 + nt * 16 + lo) * EMB + kk + hi * 8;
            b[nt].u[0] = *(const u32x4*)p;
            b[nt].u[1] = *(const u32x4*)(p + 16);
        }
#pragma unroll
        for (int mt = 0; mt < 2; ++mt)
#pragma unroll
            for (int nt = 0; nt < 4; ++nt)
                acc[mt][nt] = __builtin_amdgcn_wmma_f32_16x16x32_bf16(
                    false, a[mt].v, false, b[nt].v, (short)0, acc[mt][nt],
                    false, false);
    }

#pragma unroll
    for (int mt = 0; mt < 2; ++mt)
#pragma unroll
        for (int nt = 0; nt < 4; ++nt)
#pragma unroll
            for (int r = 0; r < 8; ++r) {
                int row = m0 + mt * 16 + hi * 8 + r;
                int e   = n0 + nt * 16 + lo;
                out[(size_t)row * EMB + e] = acc[mt][nt][r] + bv[nt];
            }
}

// ---------------------------------------------------------------------------
// Host launcher
// ---------------------------------------------------------------------------
extern "C" void kernel_launch(void* const* d_in, const int* in_sizes, int n_in,
                              void* d_out, int out_size, void* d_ws,
                              size_t ws_size, hipStream_t stream) {
    const float* query = (const float*)d_in[0];
    const float* q_w   = (const float*)d_in[1];
    const float* q_b   = (const float*)d_in[2];
    const float* k_w   = (const float*)d_in[3];
    const float* k_b   = (const float*)d_in[4];
    const float* v_w   = (const float*)d_in[5];
    const float* v_b   = (const float*)d_in[6];
    const float* out_w = (const float*)d_in[7];
    const float* out_b = (const float*)d_in[8];

    const size_t SZ_X  = (size_t)NROWS * EMB;                    // 4 Mi elems
    const size_t SZ_W  = (size_t)EMB * EMB;                      // 1 Mi elems
    const size_t SZ_HD = (size_t)BATCH * HEADS * T_SEQ * DHEAD;  // 4 Mi elems

    unsigned short* ws = (unsigned short*)d_ws;
    unsigned short* X  = ws;
    unsigned short* Wq = X  + SZ_X;
    unsigned short* Wk = Wq + SZ_W;
    unsigned short* Wv = Wk + SZ_W;
    unsigned short* Wo = Wv + SZ_W;
    unsigned short* Qb = Wo + SZ_W;
    unsigned short* Kb = Qb + SZ_HD;
    unsigned short* Vt = Kb + SZ_HD;
    unsigned short* Cx = Vt + SZ_HD;

    // 1) pack fp32 -> bf16
    {
        int n = (int)SZ_X;
        pack_bf16<<<(n + 255) / 256, 256, 0, stream>>>(query, X, n);
        n = (int)SZ_W;
        pack_bf16<<<(n + 255) / 256, 256, 0, stream>>>(q_w, Wq, n);
        pack_bf16<<<(n + 255) / 256, 256, 0, stream>>>(k_w, Wk, n);
        pack_bf16<<<(n + 255) / 256, 256, 0, stream>>>(v_w, Wv, n);
        pack_bf16<<<(n + 255) / 256, 256, 0, stream>>>(out_w, Wo, n);
    }

    // 2) projections
    dim3 gg(NROWS / 32, EMB / 64);
    gemm_qkv<<<gg, 32, 0, stream>>>(X, Wq, q_b, Qb, 0);
    gemm_qkv<<<gg, 32, 0, stream>>>(X, Wk, k_b, Kb, 1);
    gemm_qkv<<<gg, 32, 0, stream>>>(X, Wv, v_b, Vt, 2);

    // 3) attention
    dim3 gf(T_SEQ / 32, BATCH * HEADS);
    flash_attn<<<gf, 32, 0, stream>>>(Qb, Kb, Vt, Cx);

    // 4) output projection (fp32)
    gemm_out<<<gg, 32, 0, stream>>>(Cx, Wo, out_b, (float*)d_out);
}
